// WindowAttention3D_79834852098337
// MI455X (gfx1250) — compile-verified, matrix operand order checked by reading, not run on hardware
//
#include <hip/hip_runtime.h>
#include <hip/hip_bf16.h>

// ---------- types ----------
typedef __attribute__((ext_vector_type(16))) __bf16 bf16x16;
typedef __attribute__((ext_vector_type(8)))  __bf16 bf16x8;
typedef __attribute__((ext_vector_type(8)))  float  f32x8;
typedef __attribute__((ext_vector_type(4)))  float  f32x4;

// ---------- problem constants ----------
#define B_TOT   4096               // windows
#define NTOK    49                 // tokens per window (7x7)
#define CDIM    512
#define NHEAD   16
#define HDIM    32                 // c per head
#define MROWS   (B_TOT * NTOK)     // 200704, multiple of 64
#define QKVCOLS (3 * CDIM)         // 1536
#define QSCALE  0.17677669529663687f  // 32^-0.5
#define APITCH  520                // LDS A-panel pitch (elems): bank-conflict free, 16B rows
#define PPITCH  72                 // LDS P pitch (elems): 144B rows, conflict-free frag reads

__device__ __constant__ int PS_TAB[9] = {-4, 1, 2, -1, 0, 3, -2, -3, 4};

static __device__ __forceinline__ int pshift(int n) {
    int i = n / 7, j = n - i * 7;
    return PS_TAB[(i % 3) * 3 + (j % 3)];
}

static __device__ __forceinline__ f32x8 zero8() {
    f32x8 z;
#pragma unroll
    for (int e = 0; e < 8; ++e) z[e] = 0.f;
    return z;
}

static __device__ __forceinline__ f32x8 wmma_bf16(bf16x16 a, bf16x16 b, f32x8 c) {
    // D = A(16x32) * B(32x16) + C, f32 accumulate
    return __builtin_amdgcn_wmma_f32_16x16x32_bf16(false, a, false, b, (short)0, c,
                                                   false, false);
}

// pack 16 bf16 A/B-fragment values from row-major bf16 memory:
// lane holds elems [k0..k0+7] and [k0+16..k0+23] of its row (p points at row+k0)
static __device__ __forceinline__ bf16x16 ldpack_bf16(const __bf16* p) {
    bf16x8 lo = *(const bf16x8*)p;
    bf16x8 hi = *(const bf16x8*)(p + 16);
    return __builtin_shufflevector(lo, hi, 0, 1, 2, 3, 4, 5, 6, 7, 8, 9, 10, 11, 12, 13,
                                   14, 15);
}

// ============================================================
// Kernel 0a: f32 -> bf16 bulk convert (8 elems/thread, n % 2048 == 0)
// ============================================================
__global__ __launch_bounds__(256) void cvt_bf16_kernel(const float* __restrict__ src,
                                                       __bf16* __restrict__ dst) {
    size_t base = ((size_t)blockIdx.x * 256 + threadIdx.x) * 8;
    f32x4 f0 = *(const f32x4*)(src + base);
    f32x4 f1 = *(const f32x4*)(src + base + 4);
    bf16x8 r;
#pragma unroll
    for (int e = 0; e < 4; ++e) {
        r[e]     = (__bf16)f0[e];
        r[e + 4] = (__bf16)f1[e];
    }
    *(bf16x8*)(dst + base) = r;
}

// ============================================================
// Kernel 0b: relative-position bias in WMMA C-fragment order.
// biasF[((h*16 + ti*4+tj)*32 + lane)*8 + r]
// row = ti*16 + r + (lane<16?0:8), col = tj*16 + (lane&15); pad = -1e30
// ============================================================
__global__ __launch_bounds__(256) void bias_frag_kernel(const float* __restrict__ table,
                                                        const int* __restrict__ relidx,
                                                        float* __restrict__ biasF) {
    int tid  = blockIdx.x * 256 + threadIdx.x;   // 65536 total
    int r    = tid & 7;
    int lane = (tid >> 3) & 31;
    int t4   = (tid >> 8) & 15;
    int h    = tid >> 12;
    int ti = t4 >> 2, tj = t4 & 3;
    int row = ti * 16 + r + ((lane < 16) ? 0 : 8);
    int col = tj * 16 + (lane & 15);
    float v = -1e30f;
    if (row < NTOK && col < NTOK) v = table[relidx[row * NTOK + col] * NHEAD + h];
    biasF[tid] = v;
}

// ============================================================
// Kernel 1: QKV GEMM (all-bf16 operands, f32 acc) with fused q-scale
// and forward patch-shift scatter. A-panel staged in LDS once per block.
// grid (3136, 6) x 256 threads; block tile 64 M x 256 N; wave: 4 Mt x 2 Nt
// ============================================================
__global__ __launch_bounds__(256) void qkv_kernel(const __bf16* __restrict__ xb,
                                                  const __bf16* __restrict__ wb,
                                                  __bf16* __restrict__ Qs,
                                                  __bf16* __restrict__ Ks,
                                                  __bf16* __restrict__ Vts) {
    __shared__ __bf16 Abuf[64 * APITCH];
    const int wv = threadIdx.x >> 5, lane = threadIdx.x & 31;
    const int l15 = lane & 15;
    const int k0  = (lane < 16) ? 0 : 8;
    const int hi8 = (lane < 16) ? 0 : 8;
    const int Mbase = blockIdx.x * 64;
    const int Ncol0 = blockIdx.y * 256 + wv * 32;

    // cooperative A-panel fill: 64 rows x 512 bf16 = 4096 16B-chunks, 16/thread
#pragma unroll
    for (int i = 0; i < 16; ++i) {
        int c   = threadIdx.x + 256 * i;
        int row = c >> 6;
        int ko  = (c & 63) * 8;
        *(bf16x8*)&Abuf[row * APITCH + ko] =
            *(const bf16x8*)(xb + (size_t)(Mbase + row) * CDIM + ko);
    }
    __syncthreads();

    const __bf16* brow[2];
#pragma unroll
    for (int tj = 0; tj < 2; ++tj)
        brow[tj] = wb + (size_t)(Ncol0 + tj * 16 + l15) * CDIM;

    f32x8 acc[4][2];
#pragma unroll
    for (int ti = 0; ti < 4; ++ti)
#pragma unroll
        for (int tj = 0; tj < 2; ++tj) acc[ti][tj] = zero8();

    for (int kk = 0; kk < CDIM; kk += 32) {
        bf16x16 af[4], bf[2];
#pragma unroll
        for (int ti = 0; ti < 4; ++ti)
            af[ti] = ldpack_bf16(&Abuf[(ti * 16 + l15) * APITCH + kk + k0]);
#pragma unroll
        for (int tj = 0; tj < 2; ++tj) bf[tj] = ldpack_bf16(brow[tj] + kk + k0);
#pragma unroll
        for (int ti = 0; ti < 4; ++ti)
#pragma unroll
            for (int tj = 0; tj < 2; ++tj)
                acc[ti][tj] = wmma_bf16(af[ti], bf[tj], acc[ti][tj]);
    }

    // epilogue: scatter with forward shift. sel/h uniform per wave.
    const int sel = Ncol0 >> 9;          // 0=q 1=k 2=v
    const int h   = (Ncol0 >> 5) & 15;
#pragma unroll
    for (int ti = 0; ti < 4; ++ti) {
#pragma unroll
        for (int tj = 0; tj < 2; ++tj) {
            const int cc = tj * 16 + l15;
#pragma unroll
            for (int r = 0; r < 8; ++r) {
                int m = Mbase + ti * 16 + r + hi8;
                int b = m / NTOK;
                int n = m - b * NTOK;
                int s  = pshift(n);
                int wn = b & 63, t = (b >> 6) & 15, bb = b >> 10;
                int b2 = (bb << 10) + (((t + s) & 15) << 6) + wn;  // forward scatter
                float v = acc[ti][tj][r];
                if (sel == 0)
                    Qs[((size_t)(b2 * NHEAD + h) * NTOK + n) * HDIM + cc] =
                        (__bf16)(v * QSCALE);
                else if (sel == 1)
                    Ks[((size_t)(b2 * NHEAD + h) * NTOK + n) * HDIM + cc] = (__bf16)v;
                else  // V stored transposed: (c x 64-padded-N)
                    Vts[((size_t)(b2 * NHEAD + h) * HDIM + cc) * 64 + n] = (__bf16)v;
            }
        }
    }
}

// ============================================================
// Kernel 2: attention. One wave per (b,h). 128 threads = 4 waves/block.
// S = Q K^T (16 WMMA) + bias, register softmax (half-wave shfl),
// P via LDS re-layout, O = P V (16 WMMA), inverse-shift scatter.
// ============================================================
__global__ __launch_bounds__(128) void attn_kernel(const __bf16* __restrict__ Qs,
                                                   const __bf16* __restrict__ Ks,
                                                   const __bf16* __restrict__ Vts,
                                                   const float* __restrict__ biasF,
                                                   __bf16* __restrict__ Ys) {
    __shared__ __bf16 Pbuf[4][64 * PPITCH];
    const int wv = threadIdx.x >> 5, lane = threadIdx.x & 31;
    const int l15 = lane & 15;
    const int k0  = (lane < 16) ? 0 : 8;
    const int hi8 = (lane < 16) ? 0 : 8;
    const int idx = blockIdx.x * 4 + wv;
    const int b = idx >> 4, h = idx & 15;

    const __bf16* qb = Qs + (size_t)(b * NHEAD + h) * NTOK * HDIM;
    const __bf16* kb = Ks + (size_t)(b * NHEAD + h) * NTOK * HDIM;
    const __bf16* vb = Vts + (size_t)(b * NHEAD + h) * HDIM * 64;

    // Q (A-frag) and K (B-frag == row loads of K) for 4 tiles each; clamp pad rows.
    bf16x16 qf[4], kf[4];
#pragma unroll
    for (int t4 = 0; t4 < 4; ++t4) {
        int nq = t4 * 16 + l15; if (nq > 48) nq = 48;
        qf[t4] = ldpack_bf16(qb + (size_t)nq * HDIM + k0);
        kf[t4] = ldpack_bf16(kb + (size_t)nq * HDIM + k0);
    }

    // S = Q K^T  (single K-step of 32)
    f32x8 S[4][4];
#pragma unroll
    for (int ti = 0; ti < 4; ++ti)
#pragma unroll
        for (int tj = 0; tj < 4; ++tj) S[ti][tj] = wmma_bf16(qf[ti], kf[tj], zero8());

    // + bias (fragment-ordered, padded -1e30)
    const float* bF = biasF + (size_t)h * 4096;
#pragma unroll
    for (int ti = 0; ti < 4; ++ti)
#pragma unroll
        for (int tj = 0; tj < 4; ++tj) {
            f32x8 bb = *(const f32x8*)(bF + (size_t)((ti * 4 + tj) * 32 + lane) * 8);
            S[ti][tj] += bb;
        }

    // register softmax: columns live across the 16-lane half (xor 1,2,4,8)
#pragma unroll
    for (int ti = 0; ti < 4; ++ti) {
        float mrow[8];
#pragma unroll
        for (int r = 0; r < 8; ++r) {
            float v = fmaxf(fmaxf(S[ti][0][r], S[ti][1][r]),
                            fmaxf(S[ti][2][r], S[ti][3][r]));
#pragma unroll
            for (int mo = 1; mo < 16; mo <<= 1) v = fmaxf(v, __shfl_xor(v, mo, 32));
            mrow[r] = v;
        }
#pragma unroll
        for (int tj = 0; tj < 4; ++tj)
#pragma unroll
            for (int r = 0; r < 8; ++r) S[ti][tj][r] = __expf(S[ti][tj][r] - mrow[r]);
#pragma unroll
        for (int r = 0; r < 8; ++r) {
            float sm = S[ti][0][r] + S[ti][1][r] + S[ti][2][r] + S[ti][3][r];
#pragma unroll
            for (int mo = 1; mo < 16; mo <<= 1) sm += __shfl_xor(sm, mo, 32);
            float inv = 1.0f / sm;
#pragma unroll
            for (int tj = 0; tj < 4; ++tj) S[ti][tj][r] *= inv;
        }
    }

    // P -> LDS (row-major, padded pitch), D-layout writes, then A-layout reads
    __bf16* pwv = &Pbuf[wv][0];
#pragma unroll
    for (int ti = 0; ti < 4; ++ti)
#pragma unroll
        for (int tj = 0; tj < 4; ++tj) {
            const int col = tj * 16 + l15;
#pragma unroll
            for (int r = 0; r < 8; ++r)
                pwv[(ti * 16 + r + hi8) * PPITCH + col] = (__bf16)S[ti][tj][r];
        }
    asm volatile("s_wait_dscnt 0x0" ::: "memory");  // same-wave LDS RAW fence

    // O = P V  (K = 64 -> two 32-steps; V read as rows of V^T)
    f32x8 O[4][2];
#pragma unroll
    for (int ti = 0; ti < 4; ++ti)
#pragma unroll
        for (int tj = 0; tj < 2; ++tj) O[ti][tj] = zero8();
#pragma unroll
    for (int ks = 0; ks < 2; ++ks) {
        bf16x16 pf[4], vf[2];
#pragma unroll
        for (int ti = 0; ti < 4; ++ti)
            pf[ti] = ldpack_bf16(pwv + (ti * 16 + l15) * PPITCH + ks * 32 + k0);
#pragma unroll
        for (int tj = 0; tj < 2; ++tj)
            vf[tj] = ldpack_bf16(vb + (size_t)(tj * 16 + l15) * 64 + ks * 32 + k0);
#pragma unroll
        for (int ti = 0; ti < 4; ++ti)
#pragma unroll
            for (int tj = 0; tj < 2; ++tj)
                O[ti][tj] = wmma_bf16(pf[ti], vf[tj], O[ti][tj]);
    }

    // inverse-shift scatter into (B_, 49, 512) bf16
    const int wn = b & 63, t = (b >> 6) & 15, bb = b >> 10;
#pragma unroll
    for (int ti = 0; ti < 4; ++ti)
#pragma unroll
        for (int tj = 0; tj < 2; ++tj) {
            const int cc = tj * 16 + l15;
#pragma unroll
            for (int r = 0; r < 8; ++r) {
                int n = ti * 16 + r + hi8;
                if (n < NTOK) {
                    int s  = pshift(n);
                    int b2 = (bb << 10) + (((t - s) & 15) << 6) + wn;  // inverse scatter
                    Ys[((size_t)b2 * NTOK + n) * CDIM + h * HDIM + cc] =
                        (__bf16)O[ti][tj][r];
                }
            }
        }
}

// ============================================================
// Kernel 3: projection GEMM, all-bf16 operands -> f32 + bias.
// A-panel staged in LDS. grid (3136, 2) x 256 threads
// ============================================================
__global__ __launch_bounds__(256) void proj_kernel(const __bf16* __restrict__ Ys,
                                                   const __bf16* __restrict__ wb,
                                                   const float* __restrict__ pb,
                                                   float* __restrict__ out) {
    __shared__ __bf16 Abuf[64 * APITCH];
    const int wv = threadIdx.x >> 5, lane = threadIdx.x & 31;
    const int l15 = lane & 15;
    const int k0  = (lane < 16) ? 0 : 8;
    const int hi8 = (lane < 16) ? 0 : 8;
    const int Mbase = blockIdx.x * 64;
    const int Ncol0 = blockIdx.y * 256 + wv * 32;

#pragma unroll
    for (int i = 0; i < 16; ++i) {
        int c   = threadIdx.x + 256 * i;
        int row = c >> 6;
        int ko  = (c & 63) * 8;
        *(bf16x8*)&Abuf[row * APITCH + ko] =
            *(const bf16x8*)(Ys + (size_t)(Mbase + row) * CDIM + ko);
    }
    __syncthreads();

    const __bf16* brow[2];
#pragma unroll
    for (int tj = 0; tj < 2; ++tj)
        brow[tj] = wb + (size_t)(Ncol0 + tj * 16 + l15) * CDIM;

    f32x8 acc[4][2];
#pragma unroll
    for (int ti = 0; ti < 4; ++ti)
#pragma unroll
        for (int tj = 0; tj < 2; ++tj) acc[ti][tj] = zero8();

    for (int kk = 0; kk < CDIM; kk += 32) {
        bf16x16 af[4], bf[2];
#pragma unroll
        for (int ti = 0; ti < 4; ++ti)
            af[ti] = ldpack_bf16(&Abuf[(ti * 16 + l15) * APITCH + kk + k0]);
#pragma unroll
        for (int tj = 0; tj < 2; ++tj) bf[tj] = ldpack_bf16(brow[tj] + kk + k0);
#pragma unroll
        for (int ti = 0; ti < 4; ++ti)
#pragma unroll
            for (int tj = 0; tj < 2; ++tj)
                acc[ti][tj] = wmma_bf16(af[ti], bf[tj], acc[ti][tj]);
    }

#pragma unroll
    for (int tj = 0; tj < 2; ++tj) {
        const int jcol = Ncol0 + tj * 16 + l15;
        const float bias = pb[jcol];
#pragma unroll
        for (int ti = 0; ti < 4; ++ti)
#pragma unroll
            for (int r = 0; r < 8; ++r) {
                int m = Mbase + ti * 16 + r + hi8;
                out[(size_t)m * CDIM + jcol] = acc[ti][tj][r] + bias;
            }
    }
}

// ============================================================
// launch
// ============================================================
extern "C" void kernel_launch(void* const* d_in, const int* in_sizes, int n_in,
                              void* d_out, int out_size, void* d_ws, size_t ws_size,
                              hipStream_t stream) {
    const float* x      = (const float*)d_in[0];
    const float* qkv_w  = (const float*)d_in[1];
    const float* proj_w = (const float*)d_in[2];
    const float* proj_b = (const float*)d_in[3];
    const float* rtab   = (const float*)d_in[4];
    const int*   ridx   = (const int*)d_in[5];
    float* out = (float*)d_out;

    // workspace carve (bytes)
    const size_t SZ_BIASF = 65536u * 4u;                               // 256 KB
    const size_t SZ_WQ    = (size_t)QKVCOLS * CDIM * 2;                // 1.5 MB
    const size_t SZ_WP    = (size_t)CDIM * CDIM * 2;                   // 0.5 MB
    const size_t SZ_XY    = (size_t)MROWS * CDIM * 2;                  // 205.5 MB
    const size_t SZ_QK    = (size_t)B_TOT * NHEAD * NTOK * HDIM * 2;   // 205.5 MB each
    const size_t SZ_VT    = (size_t)B_TOT * NHEAD * HDIM * 64 * 2;     // 256 MB
    char* ws = (char*)d_ws;
    float*  biasF = (float*)(ws);
    __bf16* wqb   = (__bf16*)(ws + ((SZ_BIASF + 255) & ~(size_t)255));
    __bf16* wpb   = (__bf16*)((char*)wqb + SZ_WQ);
    __bf16* XY    = (__bf16*)((char*)wpb + SZ_WP);  // xb, later reused as Ys
    __bf16* Qs    = (__bf16*)((char*)XY + SZ_XY);
    __bf16* Ks    = (__bf16*)((char*)Qs + SZ_QK);
    __bf16* Vts   = (__bf16*)((char*)Ks + SZ_QK);
    (void)in_sizes; (void)n_in; (void)out_size; (void)ws_size;

    bias_frag_kernel<<<256, 256, 0, stream>>>(rtab, ridx, biasF);
    cvt_bf16_kernel<<<(MROWS * CDIM) / 2048, 256, 0, stream>>>(x, XY);
    cvt_bf16_kernel<<<(QKVCOLS * CDIM) / 2048, 256, 0, stream>>>(qkv_w, wqb);
    cvt_bf16_kernel<<<(CDIM * CDIM) / 2048, 256, 0, stream>>>(proj_w, wpb);
    qkv_kernel<<<dim3(MROWS / 64, QKVCOLS / 256), 256, 0, stream>>>(XY, wqb, Qs, Ks, Vts);
    attn_kernel<<<(B_TOT * NHEAD) / 4, 128, 0, stream>>>(Qs, Ks, Vts, biasF, XY);
    proj_kernel<<<dim3(MROWS / 64, CDIM / 256), 256, 0, stream>>>(XY, wpb, proj_b, out);
}
